// IrregularityAwareAttention_21629455303098
// MI455X (gfx1250) — matrix operand order repaired and checked by reading, not compiled, not run on hardware
//
#include <hip/hip_runtime.h>
#include <stdint.h>

// ---------------------------------------------------------------------------
// IrregularityAwareAttention for MI455X (gfx1250, wave32, WMMA).
// B=8, N=2048, D=512, DA=64.
//   k0 (x6): one-shot f32 -> bf16 conversion of x, Wq, Wk, Wv, q_aux, k_aux
//   k1: QKV projection GEMM: x tile staged in LDS, W fragments double-buffered
//       from global, inner loop = ds_load_b128 + global_load_b128 + WMMA.
//       V stored transposed [B][512][N] for contiguous PV B-fragments.
//   k2: fused flash attention: Q/Qaux tiles staged in LDS once per block,
//       K/V fragments software-pipelined, adjacency loaded ahead of the
//       score WMMAs, online softmax with 16-lane shuffle reductions.
// Workspace (bf16 elements):
//   Qb | Kb | Vt | Xb (4 * 8*2048*512) | Wb (3*512*512) | Qab | Kab (2*8*2048*64)
// ---------------------------------------------------------------------------

#define BATCH  8
#define SEQ    2048
#define DMODEL 512
#define DAUX   64

typedef __bf16 bf16_t;
typedef bf16_t v16bf __attribute__((ext_vector_type(16)));
typedef float  v8f   __attribute__((ext_vector_type(8)));

union Frag16 { v16bf v; uint4 q[2]; bf16_t h[16]; };
union Pack8  { uint4 q; bf16_t h[8]; };
union Pack4  { uint2 u; bf16_t h[4]; };

#define WMMA_BF16(A, B, C) \
  __builtin_amdgcn_wmma_f32_16x16x32_bf16(false, (A).v, false, (B).v, (short)0, (C), false, false)

// ---------------------------------------------------------------------------
// Kernel 0: vectorized f32 -> bf16 convert (n must be a multiple of 4).
// ---------------------------------------------------------------------------
__global__ __launch_bounds__(256)
void cvt_kernel(const float* __restrict__ src, bf16_t* __restrict__ dst, int n) {
  int i = (blockIdx.x * 256 + threadIdx.x) * 4;
  if (i + 3 < n) {
    float4 f = *(const float4*)(src + i);
    Pack4 p;
    p.h[0] = (bf16_t)f.x; p.h[1] = (bf16_t)f.y;
    p.h[2] = (bf16_t)f.z; p.h[3] = (bf16_t)f.w;
    *(uint2*)(dst + i) = p.u;
  }
}

// ---------------------------------------------------------------------------
// Kernel 1: QKV projection.  y = x @ W^T + b  (torch Linear semantics).
// Grid: x = 12 (mat = x>>2 in {Q,K,V}, 128-col slab = x&3), y = M/16 tiles.
// Block: 64 threads = 2 waves; each wave computes a 16x64 output strip.
// ---------------------------------------------------------------------------
__global__ __launch_bounds__(64)
void qkv_kernel(const bf16_t* __restrict__ Xb, const bf16_t* __restrict__ Wb,
                const float* __restrict__ bq, const float* __restrict__ bk,
                const float* __restrict__ bv,
                bf16_t* __restrict__ Qb, bf16_t* __restrict__ Kb,
                bf16_t* __restrict__ Vt) {
  __shared__ bf16_t xsh[16 * DMODEL];      // 16 KB x row-tile, shared by waves

  const int lane = threadIdx.x & 31;
  const int wave = threadIdx.x >> 5;
  const int n    = lane & 15;
  const int half = lane >> 4;

  const int rowBase = blockIdx.y * 16;                       // flat row (b*N+seq)
  const int mat     = blockIdx.x >> 2;                       // 0=Q 1=K 2=V
  const int colBase = (blockIdx.x & 3) * 128 + wave * 64;    // in [0,512)

  // ---- cooperative stage of the x tile (16 rows are contiguous) ----
  {
    const uint4* src4 = (const uint4*)(Xb + (size_t)rowBase * DMODEL);
    uint4* dst4 = (uint4*)xsh;
    #pragma unroll
    for (int i = 0; i < 16; ++i) dst4[threadIdx.x + 64 * i] = src4[threadIdx.x + 64 * i];
  }
  __syncthreads();

  const float* __restrict__ bias = (mat == 0) ? bq : (mat == 1) ? bk : bv;
  const bf16_t* __restrict__ wmat = Wb + (size_t)mat * DMODEL * DMODEL;

  // B-fragment base: W row (colBase + t*16 + n), K run starts at half*16
  const bf16_t* wb0 = wmat + (size_t)(colBase + n) * DMODEL + half * 16;

  v8f acc[4] = {};
  Frag16 bfr[2][4];
  #pragma unroll
  for (int t = 0; t < 4; ++t) {
    const bf16_t* wr = wb0 + (size_t)t * 16 * DMODEL;
    bfr[0][t].q[0] = *(const uint4*)(wr);
    bfr[0][t].q[1] = *(const uint4*)(wr + 8);
  }

  #pragma unroll
  for (int kk = 0; kk < 16; ++kk) {
    const int cur = kk & 1, nxt = cur ^ 1;
    if (kk < 15) {                          // prefetch next K-step's W frags
      const int ko = (kk + 1) * 32;
      #pragma unroll
      for (int t = 0; t < 4; ++t) {
        const bf16_t* wr = wb0 + (size_t)t * 16 * DMODEL + ko;
        bfr[nxt][t].q[0] = *(const uint4*)(wr);
        bfr[nxt][t].q[1] = *(const uint4*)(wr + 8);
      }
    }
    const int ka_ = kk * 32 + half * 16;    // A-frag from LDS
    Frag16 a;
    a.q[0] = *(const uint4*)(xsh + n * DMODEL + ka_);
    a.q[1] = *(const uint4*)(xsh + n * DMODEL + ka_ + 8);
    #pragma unroll
    for (int t = 0; t < 4; ++t) acc[t] = WMMA_BF16(a, bfr[cur][t], acc[t]);
  }

  // ---- bias + store ----
  const int rowHalf = rowBase + half * 8;   // first of this lane-half's 8 rows
  #pragma unroll
  for (int t = 0; t < 4; ++t) {
    const int col = colBase + t * 16 + n;
    const float bb = bias[col];
    if (mat < 2) {
      bf16_t* __restrict__ dst = (mat == 0) ? Qb : Kb;
      #pragma unroll
      for (int r = 0; r < 8; ++r)
        dst[(size_t)(rowHalf + r) * DMODEL + col] = (bf16_t)(acc[t][r] + bb);
    } else {
      const int bIdx = rowBase >> 11;        // /2048
      const int seq  = rowHalf & (SEQ - 1);  // consecutive rows -> packed store
      Pack8 pk;
      #pragma unroll
      for (int r = 0; r < 8; ++r) pk.h[r] = (bf16_t)(acc[t][r] + bb);
      *(uint4*)(Vt + ((size_t)(bIdx * DMODEL + col)) * SEQ + seq) = pk.q;
    }
  }
}

// ---------------------------------------------------------------------------
// Kernel 2: fused flash attention with aux-score merge + adjacency mask.
// Grid: (N/16 row tiles, B).  Block: 64 threads = 2 waves.
// Both waves compute the 16x32 probability block; wave w accumulates the
// 256-wide output slice e in [w*256, w*256+256)  ->  O = 16 v8f = 128 VGPRs.
// ---------------------------------------------------------------------------
__global__ __launch_bounds__(64)
void attn_kernel(const bf16_t* __restrict__ Qb, const bf16_t* __restrict__ Kb,
                 const bf16_t* __restrict__ Vt,
                 const bf16_t* __restrict__ Qa, const bf16_t* __restrict__ Ka,
                 const int* __restrict__ adj,
                 const float* __restrict__ cPtr, const float* __restrict__ thrPtr,
                 float* __restrict__ out) {
  __shared__ bf16_t qsh[16 * DMODEL];    // 16 KB Q row-tile
  __shared__ bf16_t qash[16 * DAUX];     //  2 KB aux-Q row-tile
  __shared__ bf16_t psh[2][16 * 32];     //  2 KB per-wave P transpose staging

  const int lane = threadIdx.x & 31;
  const int wave = threadIdx.x >> 5;
  const int n    = lane & 15;
  const int half = lane >> 4;
  const int b       = blockIdx.y;
  const int rowBase = blockIdx.x * 16;

  // ---- cooperative stage of Q / Qaux tiles (rows contiguous in global) ----
  {
    const uint4* src4 = (const uint4*)(Qb + (size_t)(b * SEQ + rowBase) * DMODEL);
    uint4* dst4 = (uint4*)qsh;
    #pragma unroll
    for (int i = 0; i < 16; ++i) dst4[threadIdx.x + 64 * i] = src4[threadIdx.x + 64 * i];
    const uint4* asrc4 = (const uint4*)(Qa + (size_t)(b * SEQ + rowBase) * DAUX);
    uint4* adst4 = (uint4*)qash;
    adst4[threadIdx.x * 2]     = asrc4[threadIdx.x * 2];
    adst4[threadIdx.x * 2 + 1] = asrc4[threadIdx.x * 2 + 1];
  }
  __syncthreads();

  const float c    = cPtr[0];
  const float thr  = thrPtr[0];
  const float sD   = 0.04419417382415922f;   // 1/sqrt(512)
  const float sA   = 0.125f;                 // 1/sqrt(64)
  const float NEG  = -3.0e38f;

  const int eBase = wave * 256;
  const bf16_t* __restrict__ qls  = qsh  + n * DMODEL;   // A-frag row in LDS
  const bf16_t* __restrict__ qals = qash + n * DAUX;

  v8f O[16] = {};
  float rm[8], rs[8];
  #pragma unroll
  for (int r = 0; r < 8; ++r) { rm[r] = NEG; rs[r] = 0.0f; }

  for (int it = 0; it < SEQ / 32; ++it) {
    const int colBase = it * 32;

    // ---- adjacency for this 16x32 tile: load ahead of the score WMMAs ----
    const int* __restrict__ arow =
        adj + (size_t)(b * SEQ + rowBase + half * 8) * SEQ + colBase;
    int ad0[8], ad1[8];
    #pragma unroll
    for (int r = 0; r < 8; ++r) {
      ad0[r] = arow[(size_t)r * SEQ + n];
      ad1[r] = arow[(size_t)r * SEQ + 16 + n];
    }

    // ---------------- scores = Q K^T over D=512 (pipelined) ----------------
    v8f s0 = {}, s1 = {};
    const bf16_t* kbase0 = Kb + (size_t)(b * SEQ + colBase + n) * DMODEL + half * 16;
    const bf16_t* kbase1 = kbase0 + (size_t)16 * DMODEL;
    Frag16 kf0[2], kf1[2];
    kf0[0].q[0] = *(const uint4*)(kbase0);
    kf0[0].q[1] = *(const uint4*)(kbase0 + 8);
    kf1[0].q[0] = *(const uint4*)(kbase1);
    kf1[0].q[1] = *(const uint4*)(kbase1 + 8);
    #pragma unroll
    for (int kk = 0; kk < 16; ++kk) {
      const int cur = kk & 1, nxt = cur ^ 1;
      if (kk < 15) {
        const int ko = (kk + 1) * 32;
        kf0[nxt].q[0] = *(const uint4*)(kbase0 + ko);
        kf0[nxt].q[1] = *(const uint4*)(kbase0 + ko + 8);
        kf1[nxt].q[0] = *(const uint4*)(kbase1 + ko);
        kf1[nxt].q[1] = *(const uint4*)(kbase1 + ko + 8);
      }
      const int ka_ = kk * 32 + half * 16;
      Frag16 aq;
      aq.q[0] = *(const uint4*)(qls + ka_);
      aq.q[1] = *(const uint4*)(qls + ka_ + 8);
      s0 = WMMA_BF16(aq, kf0[cur], s0);
      s1 = WMMA_BF16(aq, kf1[cur], s1);
    }

    // ---------------- aux scores over DA=64 ----------------
    v8f x0 = {}, x1 = {};
    {
      const bf16_t* kabase0 = Ka + (size_t)(b * SEQ + colBase + n) * DAUX + half * 16;
      const bf16_t* kabase1 = kabase0 + (size_t)16 * DAUX;
      Frag16 g0[2], g1[2];
      #pragma unroll
      for (int kk = 0; kk < 2; ++kk) {
        g0[kk].q[0] = *(const uint4*)(kabase0 + kk * 32);
        g0[kk].q[1] = *(const uint4*)(kabase0 + kk * 32 + 8);
        g1[kk].q[0] = *(const uint4*)(kabase1 + kk * 32);
        g1[kk].q[1] = *(const uint4*)(kabase1 + kk * 32 + 8);
      }
      #pragma unroll
      for (int kk = 0; kk < 2; ++kk) {
        const int ka_ = kk * 32 + half * 16;
        Frag16 aq;
        aq.q[0] = *(const uint4*)(qals + ka_);
        aq.q[1] = *(const uint4*)(qals + ka_ + 8);
        x0 = WMMA_BF16(aq, g0[kk], x0);
        x1 = WMMA_BF16(aq, g1[kk], x1);
      }
    }

    // -------- merge, adjacency mask, online softmax (row = r + 8*half) -----
    float alpha[8], p0v[8], p1v[8];
    #pragma unroll
    for (int r = 0; r < 8; ++r) {
      float v0 = s0[r] * sD, v1 = s1[r] * sD;
      float a0 = x0[r] * sA, a1 = x1[r] * sA;
      if (a0 != 0.0f && v0 > thr) v0 = (1.0f - c) * v0 + c * a0;
      if (a1 != 0.0f && v1 > thr) v1 = (1.0f - c) * v1 + c * a1;
      if (ad0[r] == 0) v0 = NEG;
      if (ad1[r] == 0) v1 = NEG;

      float tmax = fmaxf(v0, v1);                 // reduce over 16-lane half
      #pragma unroll
      for (int off = 1; off < 16; off <<= 1)
        tmax = fmaxf(tmax, __shfl_xor(tmax, off, 32));
      const float rmn = fmaxf(rm[r], tmax);
      alpha[r] = __expf(rm[r] - rmn);
      rm[r] = rmn;
      const float p0 = __expf(v0 - rmn);
      const float p1 = __expf(v1 - rmn);
      float ps = p0 + p1;
      #pragma unroll
      for (int off = 1; off < 16; off <<= 1) ps += __shfl_xor(ps, off, 32);
      rs[r] = rs[r] * alpha[r] + ps;
      p0v[r] = p0; p1v[r] = p1;
    }

    #pragma unroll
    for (int t = 0; t < 16; ++t)
      #pragma unroll
      for (int r = 0; r < 8; ++r) O[t][r] *= alpha[r];

    // -------- C-layout -> A-layout transpose of P through wave-private LDS --
    bf16_t* __restrict__ ps_ = psh[wave];
    #pragma unroll
    for (int r = 0; r < 8; ++r) {
      const int m = r + half * 8;
      ps_[m * 32 + n]      = (bf16_t)p0v[r];
      ps_[m * 32 + 16 + n] = (bf16_t)p1v[r];
    }
    asm volatile("s_wait_dscnt 0" ::: "memory");
    Frag16 pf;
    {
      const int k0 = half * 16;                // A-frag: row n, K runs
      pf.q[0] = *(const uint4*)(ps_ + n * 32 + k0);
      pf.q[1] = *(const uint4*)(ps_ + n * 32 + k0 + 8);
    }

    // ---------------- O += P @ V (pipelined V^T fragment loads) ------------
    const bf16_t* vbase =
        Vt + (size_t)(b * DMODEL + eBase + n) * SEQ + colBase + half * 16;
    Frag16 vf[2];
    vf[0].q[0] = *(const uint4*)(vbase);
    vf[0].q[1] = *(const uint4*)(vbase + 8);
    #pragma unroll
    for (int t = 0; t < 16; ++t) {
      const int cur = t & 1, nxt = cur ^ 1;
      if (t < 15) {
        const bf16_t* vr = vbase + (size_t)(t + 1) * 16 * SEQ;
        vf[nxt].q[0] = *(const uint4*)(vr);
        vf[nxt].q[1] = *(const uint4*)(vr + 8);
      }
      O[t] = WMMA_BF16(pf, vf[cur], O[t]);
    }
  }

  // ---------------- normalize + store ----------------
  #pragma unroll
  for (int r = 0; r < 8; ++r) rs[r] = 1.0f / rs[r];
  float* __restrict__ orow =
      out + (size_t)(b * SEQ + rowBase + half * 8) * DMODEL + eBase + n;
  #pragma unroll
  for (int t = 0; t < 16; ++t)
    #pragma unroll
    for (int r = 0; r < 8; ++r)
      orow[(size_t)r * DMODEL + t * 16] = O[t][r] * rs[r];
}

// ---------------------------------------------------------------------------
extern "C" void kernel_launch(void* const* d_in, const int* in_sizes, int n_in,
                              void* d_out, int out_size, void* d_ws, size_t ws_size,
                              hipStream_t stream) {
  (void)in_sizes; (void)n_in; (void)out_size; (void)ws_size;
  const float* x    = (const float*)d_in[0];
  const float* qa   = (const float*)d_in[1];
  const float* ka   = (const float*)d_in[2];
  const int*   adj  = (const int*)d_in[3];
  const float* mc   = (const float*)d_in[4];
  const float* Wq   = (const float*)d_in[5];
  const float* bq   = (const float*)d_in[6];
  const float* Wk   = (const float*)d_in[7];
  const float* bk   = (const float*)d_in[8];
  const float* Wv   = (const float*)d_in[9];
  const float* bv   = (const float*)d_in[10];
  const float* thr  = (const float*)d_in[11];
  float* out = (float*)d_out;

  const size_t SZ  = (size_t)BATCH * SEQ * DMODEL;  // 8,388,608 elems
  const size_t SZW = (size_t)DMODEL * DMODEL;       //   262,144 elems
  const size_t SZA = (size_t)BATCH * SEQ * DAUX;    // 1,048,576 elems

  bf16_t* ws  = (bf16_t*)d_ws;
  bf16_t* Qb  = ws;
  bf16_t* Kb  = Qb + SZ;
  bf16_t* Vt  = Kb + SZ;
  bf16_t* Xb  = Vt + SZ;
  bf16_t* Wb  = Xb + SZ;                 // Wq | Wk | Wv concatenated
  bf16_t* Qab = Wb + 3 * SZW;
  bf16_t* Kab = Qab + SZA;

  // One-shot conversions (memory-bound; ~45 MB total read).
  cvt_kernel<<<dim3((unsigned)(SZ  / 4 / 256)), dim3(256), 0, stream>>>(x,  Xb,         (int)SZ);
  cvt_kernel<<<dim3((unsigned)(SZW / 4 / 256)), dim3(256), 0, stream>>>(Wq, Wb,         (int)SZW);
  cvt_kernel<<<dim3((unsigned)(SZW / 4 / 256)), dim3(256), 0, stream>>>(Wk, Wb + SZW,   (int)SZW);
  cvt_kernel<<<dim3((unsigned)(SZW / 4 / 256)), dim3(256), 0, stream>>>(Wv, Wb + 2*SZW, (int)SZW);
  cvt_kernel<<<dim3((unsigned)(SZA / 4 / 256)), dim3(256), 0, stream>>>(qa, Qab,        (int)SZA);
  cvt_kernel<<<dim3((unsigned)(SZA / 4 / 256)), dim3(256), 0, stream>>>(ka, Kab,        (int)SZA);

  qkv_kernel<<<dim3(12, (BATCH * SEQ) / 16), dim3(64), 0, stream>>>(
      Xb, Wb, bq, bk, bv, Qb, Kb, Vt);
  attn_kernel<<<dim3(SEQ / 16, BATCH), dim3(64), 0, stream>>>(
      Qb, Kb, Vt, Qab, Kab, adj, mc, thr, out);
}
